// GNNFF_9990093930535
// MI455X (gfx1250) — matrix operand order repaired and checked by reading, not compiled, not run on hardware
//
#include <hip/hip_runtime.h>
#include <hip/hip_bf16.h>
#include <math.h>

// ---------------------------------------------------------------------------
// GNNFF forward for MI455X (gfx1250, wave32).
// All GEMMs use v_wmma_f32_16x16x32_bf16 (f32 accumulate, bf16 operands).
// Each wave computes a 16x64 output strip (4 N-tiles): one gathered A
// fragment feeds 4 WMMAs, cutting gathered-row traffic 4x vs 1 tile/wave.
// Weights are pre-converted once per launch into a fragment-native bf16
// layout Wt[n*K + k] so each B fragment is two contiguous 16B loads.
// Segment sums use global f32 atomics; batchnorm is 2-pass column reduction.
// ---------------------------------------------------------------------------

#define DEV_ __device__ __forceinline__

constexpr int GN  = 10000;   // nodes
constexpr int GEG = 200000;  // edges
constexpr int GT  = 300000;  // triplets
constexpr int GH  = 128;     // hidden / edge channels
constexpr int GL  = 3;       // layers

typedef __attribute__((ext_vector_type(16))) __bf16 v16bf;
typedef __attribute__((ext_vector_type(8)))  __bf16 v8bf;
typedef __attribute__((ext_vector_type(8)))  float  v8f;

DEV_ float sspf(float x) {            // softplus(x) - log(2), stable
  return fmaxf(x, 0.f) + log1pf(expf(-fabsf(x))) - 0.69314718056f;
}
DEV_ float sigmf(float x) { return 1.f / (1.f + expf(-x)); }

// ------------------------------ A-row loaders ------------------------------

struct PlainLoader {                  // X[m, k] row-major f32
  const float* X; int stride;
  struct Row { const float* p;
    DEV_ float get(int k) const { return p[k]; } };
  DEV_ Row prepare(int m) const { return { X + (size_t)m * stride }; }
};

struct C1Loader {                     // concat(node[i[m]], edge[m]) : K = 256
  const float* node; const float* edge; const int* iidx;
  struct Row { const float* n; const float* e;
    DEV_ float get(int k) const { return k < GH ? n[k] : e[k - GH]; } };
  DEV_ Row prepare(int m) const {
    return { node + (size_t)iidx[m] * GH, edge + (size_t)m * GH };
  }
};

struct C2Loader {                     // node[i[m]] * node[j[m]] : K = 128
  const float* node; const int* iidx; const int* jidx;
  struct Row { const float* a; const float* b;
    DEV_ float get(int k) const { return a[k] * b[k]; } };
  DEV_ Row prepare(int m) const {
    return { node + (size_t)iidx[m] * GH, node + (size_t)jidx[m] * GH };
  }
};

struct C3Loader {                     // 5-way concat : K = 640
  const float* node; const float* edge;
  const int* ii; const int* jj; const int* kk; const int* ji; const int* kj;
  struct Row { const float* a; const float* b; const float* c;
               const float* d; const float* e;
    DEV_ float get(int k) const {
      if (k < 128) return a[k];
      if (k < 256) return b[k - 128];
      if (k < 384) return c[k - 256];
      if (k < 512) return d[k - 384];
      return e[k - 512];
    } };
  DEV_ Row prepare(int m) const {
    return { node + (size_t)ii[m] * GH, node + (size_t)jj[m] * GH,
             node + (size_t)kk[m] * GH, edge + (size_t)ji[m] * GH,
             edge + (size_t)kj[m] * GH };
  }
};

// ------------------------------ WMMA GEMM ----------------------------------
// Y[M,Ncols] = act(A[M,K] @ W[K,Ncols] + bias).
// One wave per 16x64 strip: A fragment loaded once, reused by 4 WMMAs.
// Requires Ncols % 64 == 0, K % 32 == 0, M % 16 == 0 (all hold here).
// ACT: 0 = identity, 1 = shifted softplus.
template <int ACT, class L>
__global__ __launch_bounds__(128)
void wmma_gemm(L la, const __bf16* __restrict__ Wt, const float* __restrict__ bias,
               float* __restrict__ Y, int M, int K, int Ncols)
{
  const int lane   = threadIdx.x & 31;
  const int wave   = blockIdx.x * (blockDim.x >> 5) + (threadIdx.x >> 5);
  const int strips = Ncols >> 6;
  const int total  = (M >> 4) * strips;
  if (wave >= total) return;
  const int tM = wave / strips, tS = wave % strips;
  const int half  = lane >> 4;
  const int mrow  = tM * 16 + (lane & 15);
  const int ncol0 = tS * 64 + (lane & 15);

  auto row = la.prepare(mrow);
  const __bf16* wp = Wt + (size_t)ncol0 * K + 16 * half;  // B frag base (tile 0)

  v8f acc[4] = {v8f{}, v8f{}, v8f{}, v8f{}};
  for (int k0 = 0; k0 < K; k0 += 32) {
    v16bf a;
    const int kh = k0 + 16 * half;         // A frag: two runs of 8 K-values
#pragma unroll
    for (int e = 0; e < 8; ++e) a[e]     = (__bf16)row.get(kh + e);
#pragma unroll
    for (int e = 0; e < 8; ++e) a[8 + e] = (__bf16)row.get(kh + 8 + e);

    if (k0 + 32 < K) __builtin_prefetch((const void*)(wp + k0 + 32), 0, 0);

#pragma unroll
    for (int t = 0; t < 4; ++t) {
      const v8bf* wv = (const v8bf*)(wp + (size_t)(16 * t) * K + k0);
      v8bf blo = wv[0], bhi = wv[1];
      v16bf b;
#pragma unroll
      for (int e = 0; e < 8; ++e) { b[e] = blo[e]; b[8 + e] = bhi[e]; }
      acc[t] = __builtin_amdgcn_wmma_f32_16x16x32_bf16(
          false, a, false, b, (short)0, acc[t], false, false);
    }
  }

#pragma unroll
  for (int t = 0; t < 4; ++t) {
    const int nc = ncol0 + 16 * t;
    const float bc = bias ? bias[nc] : 0.f;
#pragma unroll
    for (int r = 0; r < 8; ++r) {
      const int mo = tM * 16 + r + 8 * half;
      float v = acc[t][r] + bc;
      if (ACT == 1) v = sspf(v);
      Y[(size_t)mo * Ncols + nc] = v;
    }
  }
}

// --------------------------- small utility kernels -------------------------

__global__ void zero_kernel(float* p, long long n) {
  long long i = (long long)blockIdx.x * blockDim.x + threadIdx.x;
  if (i < n) p[i] = 0.f;
}

// W[k*Ncols+n] (f32) -> Wt[n*K+k] (bf16, fragment-native)
__global__ void convert_w(const float* __restrict__ W, __bf16* __restrict__ Wt,
                          int K, int Ncols) {
  int idx = blockIdx.x * blockDim.x + threadIdx.x;
  if (idx >= K * Ncols) return;
  int k = idx / Ncols, n = idx - k * Ncols;
  Wt[(size_t)n * K + k] = (__bf16)W[idx];
}

// h1[n,c] = ssp(We1[z[n],c] + be1[c])      (one_hot @ We1 == row gather)
__global__ void embed1_kernel(const int* __restrict__ z,
                              const float* __restrict__ We1,
                              const float* __restrict__ be1,
                              float* __restrict__ h1) {
  int idx = blockIdx.x * blockDim.x + threadIdx.x;
  if (idx >= GN * GH) return;
  int n = idx >> 7, c = idx & 127;
  h1[idx] = sspf(We1[(size_t)z[n] * GH + c] + be1[c]);
}

__global__ void geom_kernel(const float* __restrict__ pos,
                            const int* __restrict__ ii, const int* __restrict__ jj,
                            float* __restrict__ dist, float* __restrict__ unit) {
  int r = blockIdx.x * blockDim.x + threadIdx.x;
  if (r >= GEG) return;
  int a = ii[r], b = jj[r];
  float dx = pos[3 * a]     - pos[3 * b];
  float dy = pos[3 * a + 1] - pos[3 * b + 1];
  float dz = pos[3 * a + 2] - pos[3 * b + 2];
  float d  = sqrtf(dx * dx + dy * dy + dz * dz);
  dist[r] = d;
  float inv = 1.f / d;
  unit[3 * r] = dx * inv; unit[3 * r + 1] = dy * inv; unit[3 * r + 2] = dz * inv;
}

__global__ void gauss_kernel(const float* __restrict__ dist, float* __restrict__ edge) {
  int idx = blockIdx.x * blockDim.x + threadIdx.x;
  if (idx >= GEG * GH) return;
  int r = idx >> 7, c = idx & 127;
  const float step  = 5.0f / 127.0f;
  const float coeff = -0.5f / (step * step);
  float t = dist[r] - step * (float)c;
  edge[idx] = expf(coeff * t * t);
}

// column sums / sums-of-squares; blockDim.x == C
__global__ void bn_stats(const float* __restrict__ X, int Mrows, float* __restrict__ sums) {
  int C = blockDim.x, c = threadIdx.x;
  float s = 0.f, s2 = 0.f;
  for (int r = blockIdx.x; r < Mrows; r += gridDim.x) {
    float v = X[(size_t)r * C + c];
    s += v; s2 += v * v;
  }
  atomicAdd(&sums[c], s);
  atomicAdd(&sums[C + c], s2);
}

__global__ void bn_finalize(const float* __restrict__ sums, int Mrows,
                            const float* __restrict__ g, const float* __restrict__ b,
                            float* __restrict__ scale, float* __restrict__ shift) {
  int C = blockDim.x, c = threadIdx.x;
  float mean = sums[c] / (float)Mrows;
  float var  = sums[C + c] / (float)Mrows - mean * mean;
  float sc   = g[c] * rsqrtf(var + 1e-5f);
  scale[c] = sc;
  shift[c] = b[c] - mean * sc;
}

// gate c1 and scatter-add into node aggregation
__global__ void apply_c1(const float* __restrict__ c1, const int* __restrict__ iidx,
                         const float* __restrict__ sc, const float* __restrict__ sh,
                         float* __restrict__ aggN) {
  int idx = blockIdx.x * blockDim.x + threadIdx.x;
  if (idx >= GEG * GH) return;
  int r = idx >> 7, c = idx & 127;
  const float* row = c1 + (size_t)r * 256;
  float f = row[c]       * sc[c]       + sh[c];
  float g = row[128 + c] * sc[128 + c] + sh[128 + c];
  atomicAdd(&aggN[(size_t)iidx[r] * GH + c], sigmf(f) * tanhf(g));
}

__global__ void node_update(float* __restrict__ node, const float* __restrict__ agg,
                            const float* __restrict__ sc, const float* __restrict__ sh) {
  int idx = blockIdx.x * blockDim.x + threadIdx.x;
  if (idx >= GN * GH) return;
  int c = idx & 127;
  node[idx] = tanhf(node[idx] + agg[idx] * sc[c] + sh[c]);
}

__global__ void gate_c2(const float* __restrict__ c2,
                        const float* __restrict__ sc, const float* __restrict__ sh,
                        float* __restrict__ gbuf) {
  int idx = blockIdx.x * blockDim.x + threadIdx.x;
  if (idx >= GEG * GH) return;
  int r = idx >> 7, c = idx & 127;
  const float* row = c2 + (size_t)r * 256;
  float f = row[c]       * sc[c]       + sh[c];
  float g = row[128 + c] * sc[128 + c] + sh[128 + c];
  gbuf[idx] = sigmf(f) * tanhf(g);
}

__global__ void apply_c3(const float* __restrict__ c3, const int* __restrict__ ji,
                         const float* __restrict__ sc, const float* __restrict__ sh,
                         float* __restrict__ agg3) {
  long long idx = (long long)blockIdx.x * blockDim.x + threadIdx.x;
  if (idx >= (long long)GT * GH) return;
  int r = (int)(idx >> 7), c = (int)(idx & 127);
  const float* row = c3 + (size_t)r * 256;
  float f = row[c]       * sc[c]       + sh[c];
  float g = row[128 + c] * sc[128 + c] + sh[128 + c];
  atomicAdd(&agg3[(size_t)ji[r] * GH + c], sigmf(f) * tanhf(g));
}

__global__ void edge_update(float* __restrict__ edge,
                            const float* __restrict__ c2g,
                            const float* __restrict__ s22, const float* __restrict__ h22,
                            const float* __restrict__ agg3,
                            const float* __restrict__ s32, const float* __restrict__ h32) {
  int idx = blockIdx.x * blockDim.x + threadIdx.x;
  if (idx >= GEG * GH) return;
  int c = idx & 127;
  float v = edge[idx]
          + c2g[idx]  * s22[c] + h22[c]
          + agg3[idx] * s32[c] + h32[c];
  edge[idx] = tanhf(v);
}

// final: per-edge dot(f2, Wf3) + bf3, scaled by unit, scatter-add to out[N,3]
__global__ __launch_bounds__(256)
void force_kernel(const float* __restrict__ f2, const float* __restrict__ Wf3,
                  const float* __restrict__ bf3, const float* __restrict__ unit,
                  const int* __restrict__ iidx, float* __restrict__ out) {
  int lane = threadIdx.x & 31;
  int r = blockIdx.x * (blockDim.x >> 5) + (threadIdx.x >> 5);
  if (r >= GEG) return;
  const float* row = f2 + (size_t)r * GH;
  float s = 0.f;
  for (int c = lane; c < GH; c += 32) s += row[c] * Wf3[c];
#pragma unroll
  for (int off = 16; off > 0; off >>= 1) s += __shfl_down(s, off, 32);
  s = __shfl(s, 0, 32) + bf3[0];
  if (lane < 3)
    atomicAdd(&out[(size_t)iidx[r] * 3 + lane], s * unit[(size_t)r * 3 + lane]);
}

// ------------------------------- host driver -------------------------------

extern "C" void kernel_launch(void* const* d_in, const int* in_sizes, int n_in,
                              void* d_out, int out_size, void* d_ws, size_t ws_size,
                              hipStream_t stream) {
  (void)in_sizes; (void)n_in; (void)out_size; (void)ws_size;

  const int*   z     = (const int*)  d_in[0];
  const float* pos   = (const float*)d_in[1];
  const int*   iE    = (const int*)  d_in[2];
  const int*   jE    = (const int*)  d_in[3];
  const int*   idx_i = (const int*)  d_in[4];
  const int*   idx_j = (const int*)  d_in[5];
  const int*   idx_k = (const int*)  d_in[6];
  const int*   idxji = (const int*)  d_in[7];
  const int*   idxkj = (const int*)  d_in[8];
  const float* We1 = (const float*)d_in[9];
  const float* be1 = (const float*)d_in[10];
  const float* We2 = (const float*)d_in[11];
  const float* be2 = (const float*)d_in[12];
  const float* We3 = (const float*)d_in[13];
  const float* be3 = (const float*)d_in[14];
  const float* Wc1 = (const float*)d_in[15];
  const float* bc1 = (const float*)d_in[16];
  const float* gc1 = (const float*)d_in[17];
  const float* bbc1= (const float*)d_in[18];
  const float* gn  = (const float*)d_in[19];
  const float* bn  = (const float*)d_in[20];
  const float* Wc2 = (const float*)d_in[21];
  const float* bc2 = (const float*)d_in[22];
  const float* gc2 = (const float*)d_in[23];
  const float* bbc2= (const float*)d_in[24];
  const float* Wc3 = (const float*)d_in[25];
  const float* bc3 = (const float*)d_in[26];
  const float* gc3 = (const float*)d_in[27];
  const float* bbc3= (const float*)d_in[28];
  const float* g22 = (const float*)d_in[29];
  const float* b22 = (const float*)d_in[30];
  const float* g32 = (const float*)d_in[31];
  const float* b32 = (const float*)d_in[32];
  const float* Wf1 = (const float*)d_in[33];
  const float* bf1 = (const float*)d_in[34];
  const float* Wf2 = (const float*)d_in[35];
  const float* bf2 = (const float*)d_in[36];
  const float* Wf3 = (const float*)d_in[37];
  const float* bf3 = (const float*)d_in[38];
  float* out = (float*)d_out;

  // ---- workspace carve-up ----
  char* ws = (char*)d_ws;
  size_t off = 0;
  auto alloc = [&](size_t bytes) -> void* {
    void* p = ws + off;
    off += (bytes + 255) & ~(size_t)255;
    return p;
  };
  float* node = (float*)alloc((size_t)GN  * GH * 4);
  float* h1   = (float*)alloc((size_t)GN  * GH * 4);
  float* aggN = (float*)alloc((size_t)GN  * GH * 4);
  float* edge = (float*)alloc((size_t)GEG * GH * 4);
  float* unit = (float*)alloc((size_t)GEG * 3 * 4);
  float* dist = (float*)alloc((size_t)GEG * 4);
  float* big  = (float*)alloc((size_t)GT  * 256 * 4);   // c1 / c2 / c3 raw
  float* gbuf = (float*)alloc((size_t)GEG * GH * 4);    // gated c2 ; force f1
  float* abuf = (float*)alloc((size_t)GEG * GH * 4);    // agg3 ; force f2
  float* sums = (float*)alloc(512 * 4);
  float* sA = (float*)alloc(256 * 4); float* hA = (float*)alloc(256 * 4);
  float* sB = (float*)alloc(256 * 4); float* hB = (float*)alloc(256 * 4);
  float* sC = (float*)alloc(256 * 4); float* hC = (float*)alloc(256 * 4);
  __bf16* wbf = (__bf16*)alloc((size_t)851968 * 2);     // all converted weights

  const size_t oWe2 = 0, oWe3 = 16384, oWf1 = 32768, oWf2 = 49152;
  const size_t oWc1 = 65536, oWc2 = oWc1 + 3 * 65536, oWc3 = oWc2 + 3 * 32768;

  auto zero = [&](float* p, long long n) {
    zero_kernel<<<(unsigned)((n + 255) / 256), 256, 0, stream>>>(p, n);
  };
  auto conv = [&](const float* W, __bf16* Wt, int K, int Nc) {
    convert_w<<<(K * Nc + 255) / 256, 256, 0, stream>>>(W, Wt, K, Nc);
  };
  // 16x64 strip per wave, 4 waves per block
  auto gemmGrid = [](int M, int Nc) { return ((M >> 4) * (Nc >> 6) + 3) / 4; };

  // ---- weight conversion (fragment-native bf16) ----
  conv(We2, wbf + oWe2, 128, 128);
  conv(We3, wbf + oWe3, 128, 128);
  conv(Wf1, wbf + oWf1, 128, 128);
  conv(Wf2, wbf + oWf2, 128, 128);
  for (int l = 0; l < GL; ++l) {
    conv(Wc1 + (size_t)l * 65536,  wbf + oWc1 + (size_t)l * 65536,  256, 256);
    conv(Wc2 + (size_t)l * 32768,  wbf + oWc2 + (size_t)l * 32768,  128, 256);
    conv(Wc3 + (size_t)l * 163840, wbf + oWc3 + (size_t)l * 163840, 640, 256);
  }

  // ---- atom-type embedding ----
  embed1_kernel<<<(GN * GH + 255) / 256, 256, 0, stream>>>(z, We1, be1, h1);
  wmma_gemm<1, PlainLoader><<<gemmGrid(GN, 128), 128, 0, stream>>>(
      PlainLoader{h1, 128}, wbf + oWe2, be2, aggN, GN, 128, 128);
  wmma_gemm<0, PlainLoader><<<gemmGrid(GN, 128), 128, 0, stream>>>(
      PlainLoader{aggN, 128}, wbf + oWe3, be3, node, GN, 128, 128);

  // ---- geometry + Gaussian edge features ----
  geom_kernel<<<(GEG + 255) / 256, 256, 0, stream>>>(pos, iE, jE, dist, unit);
  gauss_kernel<<<(GEG * GH + 255) / 256, 256, 0, stream>>>(dist, edge);

  // ---- message-passing layers ----
  for (int l = 0; l < GL; ++l) {
    // NodeUpdate: c1 = concat(node[i], edge) @ Wc1 + bc1
    wmma_gemm<0, C1Loader><<<gemmGrid(GEG, 256), 128, 0, stream>>>(
        C1Loader{node, edge, iE}, wbf + oWc1 + (size_t)l * 65536,
        bc1 + (size_t)l * 256, big, GEG, 256, 256);
    zero(sums, 512);
    bn_stats<<<1024, 256, 0, stream>>>(big, GEG, sums);
    bn_finalize<<<1, 256, 0, stream>>>(sums, GEG, gc1 + (size_t)l * 256,
                                       bbc1 + (size_t)l * 256, sA, hA);
    zero(aggN, (long long)GN * GH);
    apply_c1<<<(GEG * GH + 255) / 256, 256, 0, stream>>>(big, iE, sA, hA, aggN);
    zero(sums, 512);
    bn_stats<<<1024, 128, 0, stream>>>(aggN, GN, sums);
    bn_finalize<<<1, 128, 0, stream>>>(sums, GN, gn + (size_t)l * 128,
                                       bn + (size_t)l * 128, sA, hA);
    node_update<<<(GN * GH + 255) / 256, 256, 0, stream>>>(node, aggN, sA, hA);

    // EdgeUpdate pair term: c2 = (node[i]*node[j]) @ Wc2 + bc2
    wmma_gemm<0, C2Loader><<<gemmGrid(GEG, 256), 128, 0, stream>>>(
        C2Loader{node, iE, jE}, wbf + oWc2 + (size_t)l * 32768,
        bc2 + (size_t)l * 256, big, GEG, 128, 256);
    zero(sums, 512);
    bn_stats<<<1024, 256, 0, stream>>>(big, GEG, sums);
    bn_finalize<<<1, 256, 0, stream>>>(sums, GEG, gc2 + (size_t)l * 256,
                                       bbc2 + (size_t)l * 256, sA, hA);
    gate_c2<<<(GEG * GH + 255) / 256, 256, 0, stream>>>(big, sA, hA, gbuf);
    zero(sums, 512);
    bn_stats<<<1024, 128, 0, stream>>>(gbuf, GEG, sums);
    bn_finalize<<<1, 128, 0, stream>>>(sums, GEG, g22 + (size_t)l * 128,
                                       b22 + (size_t)l * 128, sB, hB);

    // EdgeUpdate triplet term: c3 = concat(5 gathers) @ Wc3 + bc3
    wmma_gemm<0, C3Loader><<<gemmGrid(GT, 256), 128, 0, stream>>>(
        C3Loader{node, edge, idx_i, idx_j, idx_k, idxji, idxkj},
        wbf + oWc3 + (size_t)l * 163840, bc3 + (size_t)l * 256, big, GT, 640, 256);
    zero(sums, 512);
    bn_stats<<<1024, 256, 0, stream>>>(big, GT, sums);
    bn_finalize<<<1, 256, 0, stream>>>(sums, GT, gc3 + (size_t)l * 256,
                                       bbc3 + (size_t)l * 256, sA, hA);
    zero(abuf, (long long)GEG * GH);
    apply_c3<<<(unsigned)(((long long)GT * GH + 255) / 256), 256, 0, stream>>>(
        big, idxji, sA, hA, abuf);
    zero(sums, 512);
    bn_stats<<<1024, 128, 0, stream>>>(abuf, GEG, sums);
    bn_finalize<<<1, 128, 0, stream>>>(sums, GEG, g32 + (size_t)l * 128,
                                       b32 + (size_t)l * 128, sC, hC);
    edge_update<<<(GEG * GH + 255) / 256, 256, 0, stream>>>(edge, gbuf, sB, hB,
                                                            abuf, sC, hC);
  }

  // ---- force predictor ----
  wmma_gemm<1, PlainLoader><<<gemmGrid(GEG, 128), 128, 0, stream>>>(
      PlainLoader{edge, 128}, wbf + oWf1, bf1, gbuf, GEG, 128, 128);
  wmma_gemm<1, PlainLoader><<<gemmGrid(GEG, 128), 128, 0, stream>>>(
      PlainLoader{gbuf, 128}, wbf + oWf2, bf2, abuf, GEG, 128, 128);
  zero(out, (long long)GN * 3);
  force_kernel<<<(GEG + 7) / 8, 256, 0, stream>>>(abuf, Wf3, bf3, unit, iE, out);
}
